// BatchedAttn_Net_Gated_62637803045044
// MI455X (gfx1250) — matrix operand order repaired and checked by reading, not compiled, not run on hardware
//
#include <hip/hip_runtime.h>
#include <hip/hip_bf16.h>
#include <math.h>

// Problem constants (fixed by the reference).
#define BB_   8
#define NN_   8192
#define LL_   1024
#define DD_   256

typedef __attribute__((ext_vector_type(16))) _Float16 v16h;
typedef __attribute__((ext_vector_type(8)))  _Float16 v8h;
typedef __attribute__((ext_vector_type(8)))  float    v8f;

typedef union { v16h v; v8h h[2]; } u16h;

// ---------------------------------------------------------------------------
// Kernel 0: convert Wa/Wb (f32, [L][D] row-major) to f16 transposed [D][L].
// Lets each WMMA B-fragment lane read 16 consecutive K values contiguously.
// ---------------------------------------------------------------------------
__global__ void wprep_kernel(const float* __restrict__ Wa,
                             const float* __restrict__ Wb,
                             _Float16* __restrict__ waT,
                             _Float16* __restrict__ wbT) {
    int idx = blockIdx.x * 256 + threadIdx.x;
    const int total = LL_ * DD_;
    if (idx < total) {
        int k = idx >> 8;       // row in [L]
        int d = idx & 255;      // col in [D]
        waT[(size_t)d * LL_ + k] = (_Float16)Wa[idx];
    } else {
        idx -= total;
        int k = idx >> 8;
        int d = idx & 255;
        wbT[(size_t)d * LL_ + k] = (_Float16)Wb[idx];
    }
}

// ---------------------------------------------------------------------------
// Kernel 1: per-row gated score via WMMA.
//   A[row] = sum_d tanh((x Wa)_d + ba_d) * sigmoid((x Wb)_d + bb_d) * Wc_d + bc
// One WG = 64 rows of x staged to LDS as f16 (4 row-tiles of 16).
// 8 waves; wave w owns D-columns [32w, 32w+32) for BOTH Wa and Wb:
// 2 col-tiles x 2 matrices = 4 B-fragment chains, each reused across the
// 4 row-tiles -> 16 WMMAs per 32-wide K chunk per wave, with the next
// chunk's B fragments double-buffered so loads overlap the WMMAs.
// ---------------------------------------------------------------------------
#define MROWS  64
#define XPITCH 1032   // 1024 + 8 halves pad -> rows land on distinct LDS banks

__device__ __forceinline__ v16h load_bfrag(const _Float16* p) {
    u16h b;
    b.h[0] = *(const v8h*)(p);
    b.h[1] = *(const v8h*)(p + 8);
    return b.v;
}

__device__ __forceinline__ v8f wmma_f16(v16h a, v16h b, v8f c) {
    return __builtin_amdgcn_wmma_f32_16x16x32_f16(false, a, false, b,
                                                  (short)0, c, false, false);
}

__global__ void
__launch_bounds__(256)
score_kernel(const float* __restrict__ x,
             const _Float16* __restrict__ waT,
             const _Float16* __restrict__ wbT,
             const float* __restrict__ ba,
             const float* __restrict__ bb,
             const float* __restrict__ Wc,
             const float* __restrict__ bc,
             float* __restrict__ scores) {
    __shared__ _Float16 xs[MROWS * XPITCH];
    __shared__ float red[MROWS];

    const int t    = threadIdx.x;
    const int lane = t & 31;
    const int wv   = t >> 5;

    if (t < MROWS) red[t] = 0.0f;

    // ---- stage 64 rows of x into LDS as f16 -------------------------------
    {
        const int r  = t >> 2;             // 0..63 row
        const int cb = (t & 3) * 256;      // 256 cols per thread
        const float* xrow = x + ((size_t)blockIdx.x * MROWS + r) * LL_ + cb;
        _Float16* dst = &xs[r * XPITCH + cb];
        #pragma unroll
        for (int i = 0; i < 256; i += 8) {
            float4 f0 = *(const float4*)(xrow + i);
            float4 f1 = *(const float4*)(xrow + i + 4);
            v8h h;
            h[0] = (_Float16)f0.x; h[1] = (_Float16)f0.y;
            h[2] = (_Float16)f0.z; h[3] = (_Float16)f0.w;
            h[4] = (_Float16)f1.x; h[5] = (_Float16)f1.y;
            h[6] = (_Float16)f1.z; h[7] = (_Float16)f1.w;
            *(v8h*)(dst + i) = h;
        }
    }
    __syncthreads();

    // ---- WMMA main loop ----------------------------------------------------
    const int nn = lane & 15;   // column within 16-wide tile (B/C/D layout)
    const int kh = lane >> 4;   // K-half selector for B fragments
    const int m  = lane & 15;   // row for A fragments
    const int hi = lane >> 4;   // K-group selector for A fragments
    const int nb0 = wv * 32;    // this wave's first column tile base in D

    const _Float16* wa0 = waT + (size_t)(nb0 + nn)      * LL_ + kh * 16;
    const _Float16* wa1 = waT + (size_t)(nb0 + 16 + nn) * LL_ + kh * 16;
    const _Float16* wb0 = wbT + (size_t)(nb0 + nn)      * LL_ + kh * 16;
    const _Float16* wb1 = wbT + (size_t)(nb0 + 16 + nn) * LL_ + kh * 16;

    v8f accA0[4] = {{}, {}, {}, {}};
    v8f accA1[4] = {{}, {}, {}, {}};
    v8f accG0[4] = {{}, {}, {}, {}};
    v8f accG1[4] = {{}, {}, {}, {}};

    // prime the B-fragment double buffer
    v16h cA0 = load_bfrag(wa0), cA1 = load_bfrag(wa1);
    v16h cG0 = load_bfrag(wb0), cG1 = load_bfrag(wb1);

    for (int kc = 0; kc < 32; ++kc) {
        const int k0 = kc * 32;
        const int kn = (kc < 31) ? (k0 + 32) : k0;   // last iter: harmless reload

        // A fragments for the 4 row-tiles (two ds_load_b128 each)
        u16h a[4];
        #pragma unroll
        for (int rt = 0; rt < 4; ++rt) {
            const int base = (rt * 16 + m) * XPITCH + hi * 16;
            a[rt].h[0] = *(const v8h*)&xs[base + k0];
            a[rt].h[1] = *(const v8h*)&xs[base + k0 + 8];
        }

        // issue next chunk's B fragments before consuming the current ones
        v16h nA0 = load_bfrag(wa0 + kn);
        v16h nA1 = load_bfrag(wa1 + kn);
        v16h nG0 = load_bfrag(wb0 + kn);
        v16h nG1 = load_bfrag(wb1 + kn);

        #pragma unroll
        for (int rt = 0; rt < 4; ++rt) {
            accA0[rt] = wmma_f16(a[rt].v, cA0, accA0[rt]);
            accA1[rt] = wmma_f16(a[rt].v, cA1, accA1[rt]);
            accG0[rt] = wmma_f16(a[rt].v, cG0, accG0[rt]);
            accG1[rt] = wmma_f16(a[rt].v, cG1, accG1[rt]);
        }

        cA0 = nA0; cA1 = nA1; cG0 = nG0; cG1 = nG1;
    }

    // ---- epilogue: gate, scale by Wc, reduce over D ------------------------
    const float ba0 = ba[nb0 + nn],      ba1 = ba[nb0 + 16 + nn];
    const float bb0 = bb[nb0 + nn],      bb1 = bb[nb0 + 16 + nn];
    const float wc0 = Wc[nb0 + nn],      wc1 = Wc[nb0 + 16 + nn];

    #pragma unroll
    for (int rt = 0; rt < 4; ++rt) {
        float s[8];
        #pragma unroll
        for (int i = 0; i < 8; ++i) {
            float a0 = tanhf(accA0[rt][i] + ba0);
            float g0 = 1.0f / (1.0f + __expf(-(accG0[rt][i] + bb0)));
            float a1 = tanhf(accA1[rt][i] + ba1);
            float g1 = 1.0f / (1.0f + __expf(-(accG1[rt][i] + bb1)));
            s[i] = a0 * g0 * wc0 + a1 * g1 * wc1;
        }
        // reduce across the 16 lanes that share each row (xor stays in-half)
        #pragma unroll
        for (int i = 0; i < 8; ++i) {
            #pragma unroll
            for (int off = 1; off < 16; off <<= 1)
                s[i] += __shfl_xor(s[i], off, 32);
        }
        if (nn == 0) {
            #pragma unroll
            for (int i = 0; i < 8; ++i)
                atomicAdd(&red[rt * 16 + hi * 8 + i], s[i]);  // row = i + 8*hi
        }
    }
    __syncthreads();

    if (t < MROWS)
        scores[(size_t)blockIdx.x * MROWS + t] = red[t] + bc[0];
}

// ---------------------------------------------------------------------------
// Kernel 2: per-batch masked softmax; overwrite scores with normalized
// weights in place; zero d_out for the accumulation pass.
// ---------------------------------------------------------------------------
__global__ void softmax_kernel(float* __restrict__ scores,
                               const int* __restrict__ lengths,
                               float* __restrict__ out) {
    __shared__ float sm[256];
    const int b   = blockIdx.x;
    const int t   = threadIdx.x;
    const int len = lengths[b];
    float* sb = scores + (size_t)b * NN_;

    float mx = -INFINITY;
    for (int n = t; n < NN_; n += 256) {
        float v = (n < len) ? sb[n] : -INFINITY;
        mx = fmaxf(mx, v);
    }
    sm[t] = mx; __syncthreads();
    for (int o = 128; o > 0; o >>= 1) {
        if (t < o) sm[t] = fmaxf(sm[t], sm[t + o]);
        __syncthreads();
    }
    mx = sm[0]; __syncthreads();

    float sum = 0.0f;
    for (int n = t; n < NN_; n += 256)
        if (n < len) sum += __expf(sb[n] - mx);
    sm[t] = sum; __syncthreads();
    for (int o = 128; o > 0; o >>= 1) {
        if (t < o) sm[t] += sm[t + o];
        __syncthreads();
    }
    const float inv = 1.0f / sm[0];

    for (int n = t; n < NN_; n += 256) {
        float w = (n < len) ? __expf(sb[n] - mx) * inv : 0.0f;
        sb[n] = w;
    }
    for (int l = t; l < LL_; l += 256)
        out[(size_t)b * LL_ + l] = 0.0f;
}

// ---------------------------------------------------------------------------
// Kernel 3: out[b,l] = sum_n w[b,n] * x[b,n,l]
// Grid = B * (L/256) * NSPLIT; each WG handles 256 columns over a 256-row
// slice of N. Weights are staged into LDS with the gfx1250 async-to-LDS
// path (ASYNCcnt-tracked, no VGPR round-trip), drained with
// s_wait_asynccnt before the barrier. x is streamed with non-temporal
// loads (single use; keep L2 for weights/scores); f32 atomics into d_out.
// ---------------------------------------------------------------------------
#define NSPLIT 32
__global__ void
__launch_bounds__(256)
wsum_kernel(const float* __restrict__ x,
            const float* __restrict__ weights,
            float* __restrict__ out) {
    __shared__ float wsm[256];
    const int t  = threadIdx.x;
    const int bx = blockIdx.x;
    const int b      = bx >> 7;          // 128 blocks per batch
    const int rest   = bx & 127;
    const int cchunk = rest >> 5;        // 0..3  (column chunk of 256)
    const int split  = rest & 31;        // 0..31 (slice of N)
    const int col = cchunk * 256 + t;
    const int n0  = split * (NN_ / NSPLIT);

    // async copy: global weights[b, n0 + t] -> LDS wsm[t]  (4B per lane)
    {
        const float* wp = weights + (size_t)b * NN_ + n0 + t;
        unsigned lds_off = (unsigned)(size_t)(&wsm[t]);
        asm volatile("global_load_async_to_lds_b32 %0, %1, off"
                     :: "v"(lds_off), "v"(wp)
                     : "memory");
        asm volatile("s_wait_asynccnt 0" ::: "memory");
    }
    __syncthreads();

    const float* xb = x + ((size_t)b * NN_ + n0) * LL_ + col;
    float acc = 0.0f;
    #pragma unroll 4
    for (int i = 0; i < NN_ / NSPLIT; ++i) {
        if ((i & 31) == 0)
            __builtin_prefetch(xb + (size_t)(i + 32) * LL_, 0, 0);
        acc = fmaf(wsm[i], __builtin_nontemporal_load(xb + (size_t)i * LL_), acc);
    }
    atomicAdd(&out[(size_t)b * LL_ + col], acc);
}

// ---------------------------------------------------------------------------
extern "C" void kernel_launch(void* const* d_in, const int* in_sizes, int n_in,
                              void* d_out, int out_size, void* d_ws, size_t ws_size,
                              hipStream_t stream) {
    const float* x       = (const float*)d_in[0];
    const int*   lengths = (const int*)  d_in[1];
    const float* Wa      = (const float*)d_in[2];
    const float* ba      = (const float*)d_in[3];
    const float* Wb      = (const float*)d_in[4];
    const float* bb      = (const float*)d_in[5];
    const float* Wc      = (const float*)d_in[6];
    const float* bc      = (const float*)d_in[7];
    float* out = (float*)d_out;

    // Workspace: f16 transposed weights + score/weight buffer (~1.25 MB).
    _Float16* waT   = (_Float16*)d_ws;
    _Float16* wbT   = waT + (size_t)DD_ * LL_;
    float*    score = (float*)(wbT + (size_t)DD_ * LL_);

    wprep_kernel<<<(2 * LL_ * DD_) / 256, 256, 0, stream>>>(Wa, Wb, waT, wbT);
    score_kernel<<<(BB_ * NN_) / MROWS, 256, 0, stream>>>(x, waT, wbT, ba, bb, Wc, bc, score);
    softmax_kernel<<<BB_, 256, 0, stream>>>(score, lengths, out);
    wsum_kernel<<<BB_ * (LL_ / 256) * NSPLIT, 256, 0, stream>>>(x, score, out);
}
